// GINModel_23673859736037
// MI455X (gfx1250) — matrix operand order repaired and checked by reading, not compiled, not run on hardware
//
#include <hip/hip_runtime.h>
#include <hip/hip_bf16.h>

#define N_NODES   100000
#define N_EDGES   640000
#define N_GRAPHS  512
#define HID       128
#define LAYERS    4
#define MLP_HID   256
#define GDIM      5
#define PD        (HID + GDIM)     // pooled row stride = 133
#define BN_EPS    1e-5f

typedef __attribute__((ext_vector_type(8)))  float  v8f;
typedef __attribute__((ext_vector_type(16))) __bf16 v16bf;
typedef __attribute__((ext_vector_type(8)))  __bf16 v8bf;
typedef __attribute__((ext_vector_type(4)))  __bf16 v4bf;

// ---------------- edge scatter-add: agg[dst] += x[src] -----------------
__global__ __launch_bounds__(256)
void k_edge(const int* __restrict__ src, const int* __restrict__ dst,
            const float* __restrict__ x, float* __restrict__ agg)
{
    int t = blockIdx.x * blockDim.x + threadIdx.x;
    int e = t >> 5;
    if (e >= N_EDGES) return;
    int f0 = (t & 31) * 4;
    const float4 v = *(const float4*)(x + (size_t)src[e] * HID + f0);
    float* p = agg + (size_t)dst[e] * HID + f0;
    atomicAdd(p + 0, v.x);
    atomicAdd(p + 1, v.y);
    atomicAdd(p + 2, v.z);
    atomicAdd(p + 3, v.w);
}

// ---------------- hb = bf16(cur + agg) ---------------------------------
__global__ __launch_bounds__(256)
void k_combine(const float* __restrict__ cur, const float* __restrict__ agg,
               __bf16* __restrict__ hb)
{
    size_t i = ((size_t)blockIdx.x * blockDim.x + threadIdx.x) * 4;
    if (i >= (size_t)N_NODES * HID) return;
    float4 a = *(const float4*)(cur + i);
    float4 b = *(const float4*)(agg + i);
    v4bf o;
    o[0] = (__bf16)(a.x + b.x);
    o[1] = (__bf16)(a.y + b.y);
    o[2] = (__bf16)(a.z + b.z);
    o[3] = (__bf16)(a.w + b.w);
    *(v4bf*)(hb + i) = o;
}

// ---------------- wbT[n][k] = bf16(W[k][n]) ----------------------------
__global__ __launch_bounds__(256)
void k_prepw(const float* __restrict__ W, __bf16* __restrict__ wbT)
{
    int idx = blockIdx.x * blockDim.x + threadIdx.x;   // idx = n*128 + k
    if (idx >= HID * HID) return;
    int n = idx >> 7;
    int k = idx & 127;
    wbT[idx] = (__bf16)W[k * HID + n];
}

// ---------------- WMMA GEMM: C[N x 128] = A[N x 128] * W + bias --------
// A bf16 row-major; BT = W^T bf16 (BT[n][k]); one wave -> 16 rows x 128 cols.
// BT is staged once per workgroup into LDS via the CDNA5 async global->LDS
// copy path (ASYNCcnt), then all 8 waves read B fragments from LDS.
__global__ __launch_bounds__(256)
void k_gemm(const __bf16* __restrict__ A, const __bf16* __restrict__ BT,
            const float* __restrict__ bias, float* __restrict__ C, int N)
{
    __shared__ __bf16 sB[HID * HID];   // 32 KB weight tile in LDS

    // ---- async stage BT -> LDS: 8 waves x 8 instrs x 512 B = 32 KB ----
    {
        uint32_t lbase = (uint32_t)(size_t)(&sB[0]);   // flat->LDS: addr[31:0]
        uint32_t toff  = (uint32_t)threadIdx.x * 16u;
        uint64_t gbase = (uint64_t)(size_t)BT;
#pragma unroll
        for (int i = 0; i < 8; ++i) {
            uint32_t laddr = lbase + toff + (uint32_t)i * 4096u;
            uint64_t gaddr = gbase + toff + (uint64_t)i * 4096u;
            asm volatile("global_load_async_to_lds_b128 %0, %1, off"
                         :: "v"(laddr), "v"(gaddr) : "memory");
        }
        asm volatile("s_wait_asynccnt 0x0" ::: "memory");
    }
    __syncthreads();

    int wave = (blockIdx.x * blockDim.x + threadIdx.x) >> 5;
    int lane = threadIdx.x & 31;
    int row0 = wave * 16;
    if (row0 >= N) return;
    int half = lane >> 4;
    int l15  = lane & 15;

    v8f acc[8];
#pragma unroll
    for (int i = 0; i < 8; ++i) acc[i] = (v8f){0.f,0.f,0.f,0.f,0.f,0.f,0.f,0.f};

    const __bf16* Arow = A + (size_t)(row0 + l15) * HID;
    const __bf16* sBp  = &sB[0];

#pragma unroll
    for (int k0 = 0; k0 < HID; k0 += 32) {
        // A fragment: 16x32 bf16 (16 contiguous K per lane-half)
        v8bf alo = *(const v8bf*)(Arow + k0 + 16 * half);
        v8bf ahi = *(const v8bf*)(Arow + k0 + 16 * half + 8);
        v16bf a;
#pragma unroll
        for (int i = 0; i < 8; ++i) { a[i] = alo[i]; a[8 + i] = ahi[i]; }

#pragma unroll
        for (int nt = 0; nt < 8; ++nt) {
            int n = nt * 16 + l15;
            // B fragment from LDS: column n, 16 contiguous K per lane-half
            v16bf b = *(const v16bf*)(sBp + n * HID + k0 + 16 * half);
            acc[nt] = __builtin_amdgcn_wmma_f32_16x16x32_bf16(
                false, a, false, b, (short)0, acc[nt], false, false);
        }
    }

#pragma unroll
    for (int nt = 0; nt < 8; ++nt) {
        int n = nt * 16 + l15;
        float bv = bias[n];
        float* Cp = C + (size_t)row0 * HID + n;
#pragma unroll
        for (int r = 0; r < 8; ++r)
            Cp[(size_t)(r + 8 * half) * HID] = acc[nt][r] + bv;
    }
}

// ---------------- column sums / sum-of-squares -------------------------
__global__ __launch_bounds__(256)
void k_colreduce(const float* __restrict__ X, float* __restrict__ stats)
{
    int col    = threadIdx.x & 127;
    int rphase = blockIdx.x * 2 + (threadIdx.x >> 7);
    int stride = gridDim.x * 2;
    float s = 0.f, sq = 0.f;
    for (int r = rphase; r < N_NODES; r += stride) {
        float v = X[(size_t)r * HID + col];
        s += v; sq += v * v;
    }
    __shared__ float sh[512];
    sh[threadIdx.x]       = s;
    sh[256 + threadIdx.x] = sq;
    __syncthreads();
    if (threadIdx.x < 128) {
        atomicAdd(&stats[col],       sh[threadIdx.x] + sh[threadIdx.x + 128]);
        atomicAdd(&stats[128 + col], sh[256 + threadIdx.x] + sh[256 + threadIdx.x + 128]);
    }
}

// ---------------- BN (batch stats) + ReLU; optional f32/bf16 outputs ---
__global__ __launch_bounds__(256)
void k_bn_relu(const float* __restrict__ X, const float* __restrict__ stats,
               const float* __restrict__ g, const float* __restrict__ be,
               float* __restrict__ outf, __bf16* __restrict__ outb)
{
    size_t i = (size_t)blockIdx.x * blockDim.x + threadIdx.x;
    if (i >= (size_t)N_NODES * HID) return;
    int col = (int)(i & 127);
    const float invn = 1.f / (float)N_NODES;
    float m   = stats[col] * invn;
    float var = stats[128 + col] * invn - m * m;
    float sc  = rsqrtf(var + BN_EPS) * g[col];
    float sh  = be[col] - m * sc;
    float v   = fmaxf(X[i] * sc + sh, 0.f);
    if (outf) outf[i] = v;
    if (outb) outb[i] = (__bf16)v;
}

// ---------------- global add pool: pooled[batch[i]] += h[i] ------------
__global__ __launch_bounds__(256)
void k_pool(const float* __restrict__ h, const int* __restrict__ batch,
            float* __restrict__ pooled)
{
    int t = blockIdx.x * blockDim.x + threadIdx.x;
    int node = t >> 5;
    if (node >= N_NODES) return;
    int f0 = (t & 31) * 4;
    float4 v = *(const float4*)(h + (size_t)node * HID + f0);
    float* p = pooled + (size_t)batch[node] * PD + f0;
    atomicAdd(p + 0, v.x);
    atomicAdd(p + 1, v.y);
    atomicAdd(p + 2, v.z);
    atomicAdd(p + 3, v.w);
}

__global__ __launch_bounds__(256)
void k_globals(const float* __restrict__ gf, float* __restrict__ pooled)
{
    int t = blockIdx.x * blockDim.x + threadIdx.x;
    if (t >= N_GRAPHS * GDIM) return;
    int gph = t / GDIM, j = t % GDIM;
    pooled[(size_t)gph * PD + HID + j] = gf[(size_t)gph * GDIM + j];
}

// ---------------- tiny head MLP (scalar fp32, leaky relu) --------------
__global__ __launch_bounds__(256)
void k_mlp(const float* __restrict__ in, const float* __restrict__ W,
           const float* __restrict__ bias, float* __restrict__ out,
           int M, int K, int Nout, int in_stride, float slope)
{
    int t = blockIdx.x * blockDim.x + threadIdx.x;
    if (t >= M * Nout) return;
    int row = t / Nout, n = t % Nout;
    const float* ip = in + (size_t)row * in_stride;
    float acc = bias[n];
    for (int k = 0; k < K; ++k)
        acc += ip[k] * W[(size_t)k * Nout + n];
    out[t] = (acc >= 0.f) ? acc : slope * acc;
}

extern "C" void kernel_launch(void* const* d_in, const int* in_sizes, int n_in,
                              void* d_out, int out_size, void* d_ws, size_t ws_size,
                              hipStream_t stream)
{
    (void)in_sizes; (void)n_in; (void)out_size; (void)ws_size;
    const float* x    = (const float*)d_in[0];
    const int*   ei   = (const int*)d_in[1];
    const int*   bat  = (const int*)d_in[2];
    const float* gf   = (const float*)d_in[3];
    const float* W1   = (const float*)d_in[4];
    const float* b1   = (const float*)d_in[5];
    const float* g1   = (const float*)d_in[6];
    const float* be1  = (const float*)d_in[7];
    const float* W2   = (const float*)d_in[8];
    const float* b2   = (const float*)d_in[9];
    const float* g2   = (const float*)d_in[10];
    const float* be2  = (const float*)d_in[11];
    const float* g3   = (const float*)d_in[12];
    const float* be3  = (const float*)d_in[13];
    const float* hW1  = (const float*)d_in[14];
    const float* hb1  = (const float*)d_in[15];
    const float* hW2  = (const float*)d_in[16];
    const float* hb2  = (const float*)d_in[17];
    const float* hW3  = (const float*)d_in[18];
    const float* hb3  = (const float*)d_in[19];

    const int* src = ei;
    const int* dst = ei + N_EDGES;

    const size_t NH = (size_t)N_NODES * HID;
    char* w = (char*)d_ws;
    float*  h      = (float*)w;   w += NH * sizeof(float);
    float*  t      = (float*)w;   w += NH * sizeof(float);
    __bf16* hb     = (__bf16*)w;  w += NH * sizeof(__bf16);
    __bf16* wbT    = (__bf16*)w;  w += (size_t)HID * HID * sizeof(__bf16);
    float*  stats  = (float*)w;   w += 256 * sizeof(float);
    float*  pooled = (float*)w;   w += (size_t)N_GRAPHS * PD * sizeof(float);
    float*  z1     = (float*)w;   w += (size_t)N_GRAPHS * MLP_HID * sizeof(float);
    float*  z2     = (float*)w;   w += (size_t)N_GRAPHS * MLP_HID * sizeof(float);

    const int EDGE_BLKS = (N_EDGES * 32 + 255) / 256;          // 80000
    const int CMB_BLKS  = (int)((NH / 4 + 255) / 256);         // 12500
    const int GEMM_BLKS = ((N_NODES / 16) * 32 + 255) / 256;   // 782
    const int BN_BLKS   = (int)((NH + 255) / 256);             // 50000
    const int POOL_BLKS = (N_NODES * 32 + 255) / 256;          // 12500

    const float* cur = x;
    for (int l = 0; l < LAYERS; ++l) {
        // agg = segment_sum(cur[src], dst)
        hipMemsetAsync(t, 0, NH * sizeof(float), stream);
        k_edge<<<EDGE_BLKS, 256, 0, stream>>>(src, dst, cur, t);
        // hb = bf16(cur + agg)
        k_combine<<<CMB_BLKS, 256, 0, stream>>>(cur, t, hb);

        // t = hb @ W1[l] + b1[l]
        k_prepw<<<64, 256, 0, stream>>>(W1 + (size_t)l * HID * HID, wbT);
        k_gemm<<<GEMM_BLKS, 256, 0, stream>>>(hb, wbT, b1 + l * HID, t, N_NODES);
        hipMemsetAsync(stats, 0, 256 * sizeof(float), stream);
        k_colreduce<<<256, 256, 0, stream>>>(t, stats);
        k_bn_relu<<<BN_BLKS, 256, 0, stream>>>(t, stats, g1 + l * HID, be1 + l * HID,
                                               (float*)nullptr, hb);

        // t = hb @ W2[l] + b2[l]
        k_prepw<<<64, 256, 0, stream>>>(W2 + (size_t)l * HID * HID, wbT);
        k_gemm<<<GEMM_BLKS, 256, 0, stream>>>(hb, wbT, b2 + l * HID, t, N_NODES);
        hipMemsetAsync(stats, 0, 256 * sizeof(float), stream);
        k_colreduce<<<256, 256, 0, stream>>>(t, stats);
        k_bn_relu<<<BN_BLKS, 256, 0, stream>>>(t, stats, g2 + l * HID, be2 + l * HID,
                                               h, (__bf16*)nullptr);

        // outer BN + relu (in place on h)
        hipMemsetAsync(stats, 0, 256 * sizeof(float), stream);
        k_colreduce<<<256, 256, 0, stream>>>(h, stats);
        k_bn_relu<<<BN_BLKS, 256, 0, stream>>>(h, stats, g3 + l * HID, be3 + l * HID,
                                               h, (__bf16*)nullptr);
        cur = h;
    }

    // pooling + concat globals
    hipMemsetAsync(pooled, 0, (size_t)N_GRAPHS * PD * sizeof(float), stream);
    k_pool<<<POOL_BLKS, 256, 0, stream>>>(cur, bat, pooled);
    k_globals<<<(N_GRAPHS * GDIM + 255) / 256, 256, 0, stream>>>(gf, pooled);

    // head MLP
    k_mlp<<<(N_GRAPHS * MLP_HID + 255) / 256, 256, 0, stream>>>(
        pooled, hW1, hb1, z1, N_GRAPHS, PD, MLP_HID, PD, 0.01f);
    k_mlp<<<(N_GRAPHS * MLP_HID + 255) / 256, 256, 0, stream>>>(
        z1, hW2, hb2, z2, N_GRAPHS, MLP_HID, MLP_HID, MLP_HID, 0.01f);
    k_mlp<<<(N_GRAPHS + 255) / 256, 256, 0, stream>>>(
        z2, hW3, hb3, (float*)d_out, N_GRAPHS, MLP_HID, 1, MLP_HID, 1.0f);
}
